// Net_adv_14396730376571
// MI455X (gfx1250) — matrix operand order repaired and checked by reading, not compiled, not run on hardware
//
#include <hip/hip_runtime.h>

// ---------------------------------------------------------------------------
// Types
// ---------------------------------------------------------------------------
typedef __attribute__((ext_vector_type(16))) __bf16       v16bf;
typedef __attribute__((ext_vector_type(8)))  float        v8f;
typedef __attribute__((ext_vector_type(4)))  unsigned int u32x4;

// ---------------------------------------------------------------------------
// Problem constants (from reference)
// ---------------------------------------------------------------------------
static constexpr int    NN   = 10000;
static constexpr int    MPAD = 10112;            // 79 * 128
static constexpr size_t SZ_IN  = (size_t)MPAD * 1024 * 2;   // bf16 activation buf
static constexpr size_t OFF_IN0 = 0;
static constexpr size_t OFF_IN1 = OFF_IN0 + SZ_IN;
static constexpr size_t OFF_WT  = OFF_IN1 + SZ_IN;
static constexpr size_t SZ_WT   = (size_t)2048 * 1024 * 2;  // bf16 fused Wt
static constexpr size_t OFF_H   = OFF_WT + SZ_WT;
static constexpr size_t SZ_H    = (size_t)MPAD * 2048 * 4;  // f32 GEMM out
static constexpr size_t OFF_GAT = OFF_H + SZ_H;
static constexpr size_t SZ_GAT  = (size_t)MPAD * 1024 * 4;  // f32 scatter out
static constexpr size_t OFF_ALS = OFF_GAT + SZ_GAT;
static constexpr size_t OFF_ALD = OFF_ALS + 262144;
static constexpr size_t OFF_MU  = OFF_ALD + 262144;
static constexpr size_t OFF_S   = OFF_MU  + 262144;
static constexpr size_t OFF_EX  = OFF_S   + 262144;
static constexpr size_t SZ_EX   = (size_t)170000 * 6 * 4;   // exp buffer
static constexpr size_t WS_NEED = OFF_EX + SZ_EX;

// ---------------------------------------------------------------------------
// Helpers
// ---------------------------------------------------------------------------
__device__ __forceinline__ unsigned short f2bf(float f) {
  unsigned int u = __float_as_uint(f);
  u += 0x7FFFu + ((u >> 16) & 1u);          // round-to-nearest-even
  return (unsigned short)(u >> 16);
}
__device__ __forceinline__ unsigned int fordered(float f) {
  unsigned int u = __float_as_uint(f);      // monotone float->uint mapping
  return (u & 0x80000000u) ? ~u : (u | 0x80000000u);
}
__device__ __forceinline__ float funordered(unsigned int x) {
  unsigned int u = (x & 0x80000000u) ? (x ^ 0x80000000u) : ~x;
  return __uint_as_float(u);
}

#if defined(__gfx1250__) && __has_builtin(__builtin_amdgcn_global_load_async_to_lds_b128)
#define USE_ASYNC_LDS 1
#else
#define USE_ASYNC_LDS 0
#endif

#if USE_ASYNC_LDS
__device__ __forceinline__ void async_copy_b128(const void* gsrc, void* ldst) {
  // global -> LDS direct copy, tracked by ASYNCcnt (no VGPR staging).
  // Builtin prototype (from clang diagnostic): param0 = int vector_size(16)
  // in AS1 (__device__), param1 = same vector in AS3 (LDS), then imm offset,
  // imm cpol.
  typedef int v4i __attribute__((vector_size(16)));
  typedef v4i __attribute__((address_space(1))) *g4p;
  typedef v4i __attribute__((address_space(3))) *l4p;
  __builtin_amdgcn_global_load_async_to_lds_b128(
      (g4p)(uintptr_t)gsrc,
      (l4p)(unsigned int)(uintptr_t)ldst,
      0, 0);
}
__device__ __forceinline__ void wait_async0() {
#if __has_builtin(__builtin_amdgcn_s_wait_asynccnt)
  __builtin_amdgcn_s_wait_asynccnt(0);
#else
  asm volatile("s_wait_asynccnt 0" ::: "memory");
#endif
}
#endif

// ---------------------------------------------------------------------------
// fp32 -> bf16 activation conversion with zero row padding
// ---------------------------------------------------------------------------
__global__ void k_cvt_x(const float* __restrict__ x, unsigned short* __restrict__ o,
                        int rows, int K, int Mpad) {
  int idx = blockIdx.x * blockDim.x + threadIdx.x;
  if (idx >= Mpad * K) return;
  int r = idx / K;
  o[idx] = f2bf(r < rows ? x[idx] : 0.f);
}

// Fused, transposed bf16 weights: Wt[n][k] = (n<ncW ? W[k][n] : Wl[k][n-ncW]), 0-pad
__global__ void k_cvt_wt(const float* __restrict__ W, int ncW,
                         const float* __restrict__ Wl, int ncWl,
                         unsigned short* __restrict__ Wt, int K, int Npad) {
  int idx = blockIdx.x * blockDim.x + threadIdx.x;
  if (idx >= Npad * K) return;
  int n = idx / K, k = idx - n * K;
  float v = 0.f;
  if (n < ncW)              v = W [(size_t)k * ncW  + n];
  else if (n < ncW + ncWl)  v = Wl[(size_t)k * ncWl + (n - ncW)];
  Wt[idx] = f2bf(v);
}

// ---------------------------------------------------------------------------
// bf16 WMMA GEMM:  C[M][Npad] = A[M][K] * Bt[Npad][K]^T   (f32 accumulate)
// Block 256 thr (8 waves), 128x128 block tile, 32x64 wave tile, K-tile 32,
// double-buffered LDS filled by GLOBAL_LOAD_ASYNC_TO_LDS_B128 when available.
// ---------------------------------------------------------------------------
__global__ __launch_bounds__(256) void k_gemm_bf16(
    const unsigned short* __restrict__ A,
    const unsigned short* __restrict__ Bt,
    float* __restrict__ C, int M, int Npad, int K) {
  constexpr int LS = 40;                     // padded LDS row stride (elems)
  __shared__ __align__(16) unsigned short Alds[2][128 * LS];
  __shared__ __align__(16) unsigned short Blds[2][128 * LS];

  const int tid  = threadIdx.x;
  const int lane = tid & 31;
  const int wave = tid >> 5;
  const int wRow = (wave & 3) * 32;          // 4 waves along M
  const int wCol = (wave >> 2) * 64;         // 2 waves along N
  const int blockRow = blockIdx.y * 128;
  const int blockCol = blockIdx.x * 128;
  const int l16  = lane & 15;
  const int half = lane >> 4;

  v8f acc[2][4];
  v8f vzero = {0.f, 0.f, 0.f, 0.f, 0.f, 0.f, 0.f, 0.f};
#pragma unroll
  for (int i = 0; i < 2; ++i)
#pragma unroll
    for (int j = 0; j < 4; ++j) acc[i][j] = vzero;

  const unsigned short* Abase = A  + (size_t)blockRow * K;
  const unsigned short* Bbase = Bt + (size_t)blockCol * K;

  auto loadTile = [&](int kt, int buf) {
#pragma unroll
    for (int i = 0; i < 2; ++i) {            // 512 16B segments / 256 threads
      int seg = tid + i * 256;
      int row = seg >> 2;
      int sc  = (seg & 3) * 8;
      const unsigned short* ga = &Abase[(size_t)row * K + kt * 32 + sc];
      const unsigned short* gb = &Bbase[(size_t)row * K + kt * 32 + sc];
      unsigned short* la = &Alds[buf][row * LS + sc];
      unsigned short* lb = &Blds[buf][row * LS + sc];
#if USE_ASYNC_LDS
      async_copy_b128(ga, la);
      async_copy_b128(gb, lb);
#else
      *(u32x4*)la = *(const u32x4*)ga;
      *(u32x4*)lb = *(const u32x4*)gb;
#endif
    }
  };

  union Frag { v16bf v; u32x4 u[2]; };

  const int nk = K >> 5;
  loadTile(0, 0);
#if USE_ASYNC_LDS
  wait_async0();
#endif
  __syncthreads();
  for (int kt = 0; kt < nk; ++kt) {
    int buf = kt & 1;
    if (kt + 1 < nk) loadTile(kt + 1, buf ^ 1);

    Frag a[2], b[4];
#pragma unroll
    for (int mt = 0; mt < 2; ++mt) {
      // A 16x32 layout: lane<16 -> K 0-7 & 16-23 ; lane>=16 -> K 8-15 & 24-31
      const unsigned short* p = &Alds[buf][(wRow + mt * 16 + l16) * LS + half * 8];
      a[mt].u[0] = *(const u32x4*)p;
      a[mt].u[1] = *(const u32x4*)(p + 16);
    }
#pragma unroll
    for (int nt = 0; nt < 4; ++nt) {
      // B 32x16 layout: col = lane&15 ; lane<16 -> K 0-7 & 16-23 ; lane>=16 -> K 8-15 & 24-31
      const unsigned short* p = &Blds[buf][(wCol + nt * 16 + l16) * LS + half * 8];
      b[nt].u[0] = *(const u32x4*)p;
      b[nt].u[1] = *(const u32x4*)(p + 16);
    }
#pragma unroll
    for (int mt = 0; mt < 2; ++mt)
#pragma unroll
      for (int nt = 0; nt < 4; ++nt)
        acc[mt][nt] = __builtin_amdgcn_wmma_f32_16x16x32_bf16(
            false, a[mt].v, false, b[nt].v, (short)0, acc[mt][nt], false, false);
#if USE_ASYNC_LDS
    wait_async0();                           // own async fills of buf^1 done
#endif
    __syncthreads();
  }

  // C/D layout: VGPR r, lanes 0-15 -> M=r, lanes 16-31 -> M=r+8, N=lane&15
#pragma unroll
  for (int mt = 0; mt < 2; ++mt)
#pragma unroll
    for (int nt = 0; nt < 4; ++nt) {
      int row0 = blockRow + wRow + mt * 16 + half * 8;
      int col  = blockCol + wCol + nt * 16 + l16;
#pragma unroll
      for (int r = 0; r < 8; ++r)
        C[(size_t)(row0 + r) * Npad + col] = acc[mt][nt][r];
    }
}

// ---------------------------------------------------------------------------
// GAT edge phase
// ---------------------------------------------------------------------------
__global__ void k_fill_f32(float* p, float v, int n) {
  int i = blockIdx.x * blockDim.x + threadIdx.x;
  if (i < n) p[i] = v;
}
__global__ void k_fill_u32(unsigned int* p, unsigned int v, int n) {
  int i = blockIdx.x * blockDim.x + threadIdx.x;
  if (i < n) p[i] = v;
}

// per-node attention logits: alS/alD[n][h] = <h[n,h,:], a_src/a_dst[h,:]>
__global__ void k_alpha(const float* __restrict__ Hm, int ldH,
                        const float* __restrict__ as, const float* __restrict__ ad,
                        float* __restrict__ alS, float* __restrict__ alD,
                        int Nn, int Hh, int Cc) {
  int idx = blockIdx.x * blockDim.x + threadIdx.x;
  if (idx >= Nn * Hh) return;
  int n = idx / Hh, h = idx - n * Hh;
  const float* hr  = Hm + (size_t)n * ldH + h * Cc;
  const float* asr = as + h * Cc;
  const float* adr = ad + h * Cc;
  float ss = 0.f, sd = 0.f;
  for (int c = 0; c < Cc; ++c) {
    float v = hr[c];
    ss += v * asr[c];
    sd += v * adr[c];
  }
  alS[idx] = ss;
  alD[idx] = sd;
}

__device__ __forceinline__ void edge_ids(const int* ei, int E, int et,
                                         int& src, int& dst) {
  if (et < E) { src = ei[et]; dst = ei[E + et]; }
  else        { src = dst = et - E; }        // self loops appended
}

__global__ void k_edge_max(const int* __restrict__ ei, int E, int Nn,
                           const float* __restrict__ alS, const float* __restrict__ alD,
                           unsigned int* __restrict__ mU, int Hh) {
  int idx = blockIdx.x * blockDim.x + threadIdx.x;
  if (idx >= (E + Nn) * Hh) return;
  int et = idx / Hh, h = idx - et * Hh;
  int src, dst; edge_ids(ei, E, et, src, dst);
  float e = alS[src * Hh + h] + alD[dst * Hh + h];
  e = e > 0.f ? e : 0.2f * e;                // LeakyReLU(0.2)
  atomicMax(&mU[dst * Hh + h], fordered(e));
}

__global__ void k_edge_exp(const int* __restrict__ ei, int E, int Nn,
                           const float* __restrict__ alS, const float* __restrict__ alD,
                           const unsigned int* __restrict__ mU,
                           float* __restrict__ s, float* __restrict__ exb, int Hh) {
  int idx = blockIdx.x * blockDim.x + threadIdx.x;
  if (idx >= (E + Nn) * Hh) return;
  int et = idx / Hh, h = idx - et * Hh;
  int src, dst; edge_ids(ei, E, et, src, dst);
  float e = alS[src * Hh + h] + alD[dst * Hh + h];
  e = e > 0.f ? e : 0.2f * e;
  float ex = __expf(e - funordered(mU[dst * Hh + h]));
  exb[idx] = ex;
  atomicAdd(&s[dst * Hh + h], ex);
}

// out[dst,h,c] += alpha(e,h) * h[src,h,c] ; one block per edge, coalesced
__global__ __launch_bounds__(256) void k_edge_scatter(
    const int* __restrict__ ei, int E, int Nn,
    const float* __restrict__ exb, const float* __restrict__ s,
    const float* __restrict__ Hm, int ldH,
    float* __restrict__ gat, int ldG, int Hh, int Cc) {
  int et = blockIdx.x;
  int src, dst; edge_ids(ei, E, et, src, dst);
  int total = Hh * Cc;
  for (int i = threadIdx.x; i < total; i += blockDim.x) {
    int h = i / Cc;
    float alpha = exb[(size_t)et * Hh + h] / (s[(size_t)dst * Hh + h] + 1e-16f);
    atomicAdd(&gat[(size_t)dst * ldG + i], alpha * Hm[(size_t)src * ldH + i]);
  }
}

// elu(gat + b + lin + bl) -> bf16 next-layer input (zero row padding)
__global__ void k_finish12(const float* __restrict__ gat, int ldG,
                           const float* __restrict__ b,
                           const float* __restrict__ Hm, int ldH, int linOff,
                           const float* __restrict__ bl,
                           unsigned short* __restrict__ xout,
                           int Kc, int rows, int Mpad) {
  int idx = blockIdx.x * blockDim.x + threadIdx.x;
  if (idx >= Mpad * Kc) return;
  int r = idx / Kc, c = idx - r * Kc;
  float v = 0.f;
  if (r < rows) {
    v = gat[(size_t)r * ldG + c] + b[c] + Hm[(size_t)r * ldH + linOff + c] + bl[c];
    v = v > 0.f ? v : (__expf(v) - 1.f);     // ELU
  }
  xout[idx] = f2bf(v);
}

// head-mean + biases + linear + log_softmax over 6 classes
__global__ void k_final(const float* __restrict__ gat, const float* __restrict__ b3,
                        const float* __restrict__ Hm, int ldH, int linOff,
                        const float* __restrict__ bl3, float* __restrict__ out, int Nn) {
  int n = blockIdx.x * blockDim.x + threadIdx.x;
  if (n >= Nn) return;
  float z[6];
  float mx = -1e30f;
#pragma unroll
  for (int c = 0; c < 6; ++c) {
    float acc = 0.f;
#pragma unroll
    for (int h = 0; h < 6; ++h) acc += gat[(size_t)n * 36 + h * 6 + c];
    float v = acc * (1.f / 6.f) + b3[c] + Hm[(size_t)n * ldH + linOff + c] + bl3[c];
    z[c] = v;
    mx = fmaxf(mx, v);
  }
  float se = 0.f;
#pragma unroll
  for (int c = 0; c < 6; ++c) se += __expf(z[c] - mx);
  float lse = mx + __logf(se);
#pragma unroll
  for (int c = 0; c < 6; ++c) out[(size_t)n * 6 + c] = z[c] - lse;
}

// ---------------------------------------------------------------------------
// Launch
// ---------------------------------------------------------------------------
static inline dim3 g1(int n) { return dim3((n + 255) / 256); }

extern "C" void kernel_launch(void* const* d_in, const int* in_sizes, int n_in,
                              void* d_out, int out_size, void* d_ws, size_t ws_size,
                              hipStream_t stream) {
  if (n_in < 20 || ws_size < WS_NEED) return;

  const float* x   = (const float*)d_in[0];
  const int*   ei  = (const int*)  d_in[1];   // int32 (JAX x64-off)
  const float* W1  = (const float*)d_in[2];
  const float* as1 = (const float*)d_in[3];
  const float* ad1 = (const float*)d_in[4];
  const float* b1  = (const float*)d_in[5];
  const float* Wl1 = (const float*)d_in[6];
  const float* bl1 = (const float*)d_in[7];
  const float* W2  = (const float*)d_in[8];
  const float* as2 = (const float*)d_in[9];
  const float* ad2 = (const float*)d_in[10];
  const float* b2  = (const float*)d_in[11];
  const float* Wl2 = (const float*)d_in[12];
  const float* bl2 = (const float*)d_in[13];
  const float* W3  = (const float*)d_in[14];
  const float* as3 = (const float*)d_in[15];
  const float* ad3 = (const float*)d_in[16];
  const float* b3  = (const float*)d_in[17];
  const float* Wl3 = (const float*)d_in[18];
  const float* bl3 = (const float*)d_in[19];
  float* out = (float*)d_out;

  const int E    = in_sizes[1] / 2;          // 160000
  const int Etot = E + NN;
  if ((size_t)Etot * 6 * 4 > SZ_EX) return;

  char* ws = (char*)d_ws;
  unsigned short* IN0 = (unsigned short*)(ws + OFF_IN0);
  unsigned short* IN1 = (unsigned short*)(ws + OFF_IN1);
  unsigned short* WT  = (unsigned short*)(ws + OFF_WT);
  float*          Hb  = (float*)(ws + OFF_H);
  float*          GAT = (float*)(ws + OFF_GAT);
  float*          ALS = (float*)(ws + OFF_ALS);
  float*          ALD = (float*)(ws + OFF_ALD);
  unsigned int*   MU  = (unsigned int*)(ws + OFF_MU);
  float*          S   = (float*)(ws + OFF_S);
  float*          EX  = (float*)(ws + OFF_EX);

  const unsigned int NEG_INF_ORD = 0x007FFFFFu; // fordered(-inf-ish)

  // ---- Layer 1: K=512, GAT H=4 C=256, fused Ncols=2048 -------------------
  k_cvt_x  <<<g1(MPAD * 512), 256, 0, stream>>>(x, IN0, NN, 512, MPAD);
  k_cvt_wt <<<g1(2048 * 512), 256, 0, stream>>>(W1, 1024, Wl1, 1024, WT, 512, 2048);
  k_gemm_bf16<<<dim3(16, MPAD / 128), 256, 0, stream>>>(IN0, WT, Hb, MPAD, 2048, 512);
  k_alpha  <<<g1(NN * 4), 256, 0, stream>>>(Hb, 2048, as1, ad1, ALS, ALD, NN, 4, 256);
  k_fill_u32<<<g1(NN * 4), 256, 0, stream>>>(MU, NEG_INF_ORD, NN * 4);
  k_fill_f32<<<g1(NN * 4), 256, 0, stream>>>(S, 0.f, NN * 4);
  k_fill_f32<<<g1(NN * 1024), 256, 0, stream>>>(GAT, 0.f, NN * 1024);
  k_edge_max<<<g1(Etot * 4), 256, 0, stream>>>(ei, E, NN, ALS, ALD, MU, 4);
  k_edge_exp<<<g1(Etot * 4), 256, 0, stream>>>(ei, E, NN, ALS, ALD, MU, S, EX, 4);
  k_edge_scatter<<<dim3(Etot), 256, 0, stream>>>(ei, E, NN, EX, S, Hb, 2048, GAT, 1024, 4, 256);
  k_finish12<<<g1(MPAD * 1024), 256, 0, stream>>>(GAT, 1024, b1, Hb, 2048, 1024, bl1,
                                                  IN1, 1024, NN, MPAD);

  // ---- Layer 2: K=1024, GAT H=4 C=256, fused Ncols=2048 ------------------
  k_cvt_wt <<<g1(2048 * 1024), 256, 0, stream>>>(W2, 1024, Wl2, 1024, WT, 1024, 2048);
  k_gemm_bf16<<<dim3(16, MPAD / 128), 256, 0, stream>>>(IN1, WT, Hb, MPAD, 2048, 1024);
  k_alpha  <<<g1(NN * 4), 256, 0, stream>>>(Hb, 2048, as2, ad2, ALS, ALD, NN, 4, 256);
  k_fill_u32<<<g1(NN * 4), 256, 0, stream>>>(MU, NEG_INF_ORD, NN * 4);
  k_fill_f32<<<g1(NN * 4), 256, 0, stream>>>(S, 0.f, NN * 4);
  k_fill_f32<<<g1(NN * 1024), 256, 0, stream>>>(GAT, 0.f, NN * 1024);
  k_edge_max<<<g1(Etot * 4), 256, 0, stream>>>(ei, E, NN, ALS, ALD, MU, 4);
  k_edge_exp<<<g1(Etot * 4), 256, 0, stream>>>(ei, E, NN, ALS, ALD, MU, S, EX, 4);
  k_edge_scatter<<<dim3(Etot), 256, 0, stream>>>(ei, E, NN, EX, S, Hb, 2048, GAT, 1024, 4, 256);
  k_finish12<<<g1(MPAD * 1024), 256, 0, stream>>>(GAT, 1024, b2, Hb, 2048, 1024, bl2,
                                                  IN0, 1024, NN, MPAD);

  // ---- Layer 3: K=1024, GAT H=6 C=6 (mean), cols 0..35 = W3, 36..41 = Wl3
  k_cvt_wt <<<g1(128 * 1024), 256, 0, stream>>>(W3, 36, Wl3, 6, WT, 1024, 128);
  k_gemm_bf16<<<dim3(1, MPAD / 128), 256, 0, stream>>>(IN0, WT, Hb, MPAD, 128, 1024);
  k_alpha  <<<g1(NN * 6), 256, 0, stream>>>(Hb, 128, as3, ad3, ALS, ALD, NN, 6, 6);
  k_fill_u32<<<g1(NN * 6), 256, 0, stream>>>(MU, NEG_INF_ORD, NN * 6);
  k_fill_f32<<<g1(NN * 6), 256, 0, stream>>>(S, 0.f, NN * 6);
  k_fill_f32<<<g1(NN * 36), 256, 0, stream>>>(GAT, 0.f, NN * 36);
  k_edge_max<<<g1(Etot * 6), 256, 0, stream>>>(ei, E, NN, ALS, ALD, MU, 6);
  k_edge_exp<<<g1(Etot * 6), 256, 0, stream>>>(ei, E, NN, ALS, ALD, MU, S, EX, 6);
  k_edge_scatter<<<dim3(Etot), 256, 0, stream>>>(ei, E, NN, EX, S, Hb, 128, GAT, 36, 6, 6);
  k_final  <<<g1(NN), 256, 0, stream>>>(GAT, b3, Hb, 128, 36, bl3, out, NN);
}
